// Processor_60902636257602
// MI455X (gfx1250) — matrix operand order repaired and checked by reading, not compiled, not run on hardware
//
#include <hip/hip_runtime.h>
#include <stdint.h>
#include <stddef.h>

typedef __attribute__((ext_vector_type(16))) __bf16 v16bf;
typedef __attribute__((ext_vector_type(2)))  __bf16 bf16x2;
typedef __attribute__((ext_vector_type(8)))  float  v8f;
typedef __bf16 bf16;

#define DDIM 256
#define KCH 8            // 256 / 32 k-chunks per GEMM
#define TILE_M 64        // rows per workgroup
#define H1_STRIDE 264    // bf16 elements, padded (bank-conflict-free stage-2 A reads)
#define OUT_STRIDE 260   // f32 elements, padded
#define LDS_BYTES (TILE_M * OUT_STRIDE * 4)   // 66560 B; h1 (64*264*2=33792 B) overlaid

// ---- WMMA fragment packing per CDNA5 ISA layouts (wave32) ----
// A (16x32 bf16): lanes 0-15 rows M=0..15; VGPR v holds K pair; k = (v<4?2v:2v+8)+8*half
// B (32x16 bf16): lane = column N; VGPR v holds K pair; k = (v<4?2v:2v+8)+8*half
// C/D (16x16 f32): m = r + 8*half, n = lane&15

__device__ __forceinline__ v16bf frag_a_rowmajor(const bf16* __restrict__ row_ptr,
                                                 int kbase, int half) {
  v16bf a;
#pragma unroll
  for (int v = 0; v < 8; ++v) {
    int k = kbase + ((v < 4) ? (2 * v) : (2 * v + 8)) + 8 * half;
    bf16x2 p = *(const bf16x2*)(row_ptr + k);
    a[2 * v]     = p.x;
    a[2 * v + 1] = p.y;
  }
  return a;
}

__device__ __forceinline__ v16bf frag_b_from_wT(const bf16* __restrict__ w_row,
                                                int kbase, int half) {
  // w_row = W + n*256 (torch Linear weight [out,in]); B[k][n] = W[n][k]
  v16bf b;
#pragma unroll
  for (int v = 0; v < 8; ++v) {
    int k = kbase + ((v < 4) ? (2 * v) : (2 * v + 8)) + 8 * half;
    bf16x2 p = *(const bf16x2*)(w_row + k);
    b[2 * v]     = p.x;
    b[2 * v + 1] = p.y;
  }
  return b;
}

// Fused 2-layer MLP for a 64-row tile: lds_out[64][OUT_STRIDE] = relu(in@W1^T+b1)@W2^T+b2
// Wave partition: wave owns 32 rows (2 strips) x 64 cols (4 tiles); each B fragment
// is shared by 2 WMMAs -> half the B traffic per wmma vs 16x128 partition.
__device__ __forceinline__ void mlp_tile(const bf16* __restrict__ in, int row0, int Mmax,
                                         const bf16* __restrict__ W1, const float* __restrict__ b1,
                                         const bf16* __restrict__ W2, const float* __restrict__ b2,
                                         bf16* lds_h1, float* lds_out) {
  const int tid  = threadIdx.x;
  const int lane = tid & 31;
  const int wv   = tid >> 5;    // 0..7
  const int rg   = wv & 1;      // 32-row group within tile
  const int cq   = wv >> 1;     // 64-column quarter
  const int half = lane >> 4;
  const int l15  = lane & 15;

  // ---- stage 1: h1 = relu(in @ W1^T + b1) -> LDS (bf16) ----
  int ar0 = row0 + rg * 32 + l15;        // strip 0
  int ar1 = ar0 + 16;                    // strip 1
  if (ar0 >= Mmax) ar0 = Mmax - 1;       // clamp partial tiles (rows unused later)
  if (ar1 >= Mmax) ar1 = Mmax - 1;
  const bf16* arow0 = in + (size_t)ar0 * DDIM;
  const bf16* arow1 = in + (size_t)ar1 * DDIM;

  v16bf afrag[2][KCH];
#pragma unroll
  for (int kc = 0; kc < KCH; ++kc) {
    afrag[0][kc] = frag_a_rowmajor(arow0, kc * 32, half);
    afrag[1][kc] = frag_a_rowmajor(arow1, kc * 32, half);
  }

  for (int ct = 0; ct < 4; ++ct) {
    const int n0 = cq * 64 + ct * 16;
    const bf16* wrow = W1 + (size_t)(n0 + l15) * DDIM;
    v8f acc0 = {}, acc1 = {};
#pragma unroll
    for (int kc = 0; kc < KCH; ++kc) {
      v16bf b = frag_b_from_wT(wrow, kc * 32, half);
      acc0 = __builtin_amdgcn_wmma_f32_16x16x32_bf16(false, afrag[0][kc], false, b,
                                                     (short)0, acc0, false, false);
      acc1 = __builtin_amdgcn_wmma_f32_16x16x32_bf16(false, afrag[1][kc], false, b,
                                                     (short)0, acc1, false, false);
    }
    const float bias = b1[n0 + l15];
#pragma unroll
    for (int r = 0; r < 8; ++r) {
      float v0 = acc0[r] + bias;
      float v1 = acc1[r] + bias;
      v0 = v0 > 0.0f ? v0 : 0.0f;        // ReLU
      v1 = v1 > 0.0f ? v1 : 0.0f;
      const int m0 = rg * 32 + r + 8 * half;
      lds_h1[m0 * H1_STRIDE + n0 + l15]        = (bf16)v0;
      lds_h1[(m0 + 16) * H1_STRIDE + n0 + l15] = (bf16)v1;
    }
  }
  __syncthreads();

  // ---- stage 2: out = h1 @ W2^T + b2 -> LDS (f32, overlays h1 after barrier) ----
  v16bf afrag2[2][KCH];
  {
    const bf16* hrow0 = lds_h1 + (rg * 32 + l15) * H1_STRIDE;
    const bf16* hrow1 = hrow0 + 16 * H1_STRIDE;
#pragma unroll
    for (int kc = 0; kc < KCH; ++kc) {
      afrag2[0][kc] = frag_a_rowmajor(hrow0, kc * 32, half);
      afrag2[1][kc] = frag_a_rowmajor(hrow1, kc * 32, half);
    }
  }
  __syncthreads();   // h1 fully consumed; safe to overwrite with f32 out

  for (int ct = 0; ct < 4; ++ct) {
    const int n0 = cq * 64 + ct * 16;
    const bf16* wrow = W2 + (size_t)(n0 + l15) * DDIM;
    v8f acc0 = {}, acc1 = {};
#pragma unroll
    for (int kc = 0; kc < KCH; ++kc) {
      v16bf b = frag_b_from_wT(wrow, kc * 32, half);
      acc0 = __builtin_amdgcn_wmma_f32_16x16x32_bf16(false, afrag2[0][kc], false, b,
                                                     (short)0, acc0, false, false);
      acc1 = __builtin_amdgcn_wmma_f32_16x16x32_bf16(false, afrag2[1][kc], false, b,
                                                     (short)0, acc1, false, false);
    }
    const float bias = b2[n0 + l15];
#pragma unroll
    for (int r = 0; r < 8; ++r) {
      const int m0 = rg * 32 + r + 8 * half;
      lds_out[m0 * OUT_STRIDE + n0 + l15]        = acc0[r] + bias;
      lds_out[(m0 + 16) * OUT_STRIDE + n0 + l15] = acc1[r] + bias;
    }
  }
  __syncthreads();
}

// edge MLP + LN + gather x[src] + scatter-add into aggr
__global__ void __launch_bounds__(256) edge_layer_kernel(
    const bf16* __restrict__ ea, const bf16* __restrict__ W1, const float* __restrict__ b1,
    const bf16* __restrict__ W2, const float* __restrict__ b2,
    const float* __restrict__ g, const float* __restrict__ beta,
    const float* __restrict__ x, const int* __restrict__ srcs, const int* __restrict__ tgts,
    float* __restrict__ aggr, int E) {
  extern __shared__ char smem[];
  bf16*  lds_h1  = (bf16*)smem;
  float* lds_out = (float*)smem;
  const int row0 = blockIdx.x * TILE_M;

  mlp_tile(ea, row0, E, W1, b1, W2, b2, lds_h1, lds_out);

  const int lane = threadIdx.x & 31;
  const int wv   = threadIdx.x >> 5;
#pragma unroll
  for (int rr = 0; rr < 8; ++rr) {
    const int lr = wv * 8 + rr;       // wave-uniform row -> shuffles are convergent
    const int e  = row0 + lr;
    if (e >= E) continue;
    const float* orow = lds_out + lr * OUT_STRIDE;
    float s = 0.0f, q = 0.0f;
#pragma unroll
    for (int j = 0; j < 8; ++j) {
      const float v = orow[lane + 32 * j];
      s += v; q += v * v;
    }
#pragma unroll
    for (int off = 16; off >= 1; off >>= 1) {   // wave32 butterfly reduce
      s += __shfl_xor(s, off, 32);
      q += __shfl_xor(q, off, 32);
    }
    const float mu   = s * (1.0f / 256.0f);
    const float var  = q * (1.0f / 256.0f) - mu * mu;
    const float rinv = rsqrtf(var + 1e-5f);
    const int sn = srcs[e], tn = tgts[e];
    const float* xrow = x + (size_t)sn * DDIM;
    float* arow = aggr + (size_t)tn * DDIM;
#pragma unroll
    for (int j = 0; j < 8; ++j) {
      const int c = lane + 32 * j;
      const float v = (orow[c] - mu) * rinv * g[c] + beta[c];
      atomicAdd(arow + c, v + xrow[c]);         // msg = e + x[src]; scatter-add
    }
  }
}

// node MLP + LN + residual (in-place x update)
__global__ void __launch_bounds__(256) node_layer_kernel(
    const bf16* __restrict__ ain, const bf16* __restrict__ W1, const float* __restrict__ b1,
    const bf16* __restrict__ W2, const float* __restrict__ b2,
    const float* __restrict__ g, const float* __restrict__ beta,
    float* __restrict__ x, int Nn) {
  extern __shared__ char smem[];
  bf16*  lds_h1  = (bf16*)smem;
  float* lds_out = (float*)smem;
  const int row0 = blockIdx.x * TILE_M;

  mlp_tile(ain, row0, Nn, W1, b1, W2, b2, lds_h1, lds_out);

  const int lane = threadIdx.x & 31;
  const int wv   = threadIdx.x >> 5;
#pragma unroll
  for (int rr = 0; rr < 8; ++rr) {
    const int lr = wv * 8 + rr;
    const int n  = row0 + lr;
    if (n >= Nn) continue;
    const float* orow = lds_out + lr * OUT_STRIDE;
    float s = 0.0f, q = 0.0f;
#pragma unroll
    for (int j = 0; j < 8; ++j) {
      const float v = orow[lane + 32 * j];
      s += v; q += v * v;
    }
#pragma unroll
    for (int off = 16; off >= 1; off >>= 1) {
      s += __shfl_xor(s, off, 32);
      q += __shfl_xor(q, off, 32);
    }
    const float mu   = s * (1.0f / 256.0f);
    const float var  = q * (1.0f / 256.0f) - mu * mu;
    const float rinv = rsqrtf(var + 1e-5f);
    float* xrow = x + (size_t)n * DDIM;
#pragma unroll
    for (int j = 0; j < 8; ++j) {
      const int c = lane + 32 * j;
      const float v = (orow[c] - mu) * rinv * g[c] + beta[c];
      xrow[c] = v + xrow[c];                    // residual
    }
  }
}

__global__ void cvt_f32_to_bf16(const float* __restrict__ in, bf16* __restrict__ out, size_t n) {
  size_t i = (size_t)blockIdx.x * blockDim.x + threadIdx.x;
  const size_t stride = (size_t)gridDim.x * blockDim.x;
  for (; i < n; i += stride) out[i] = (bf16)in[i];
}

extern "C" void kernel_launch(void* const* d_in, const int* in_sizes, int n_in,
                              void* d_out, int out_size, void* d_ws, size_t ws_size,
                              hipStream_t stream) {
  const int NN = 100000, NE = 300000, L = 9;
  const size_t ND = (size_t)NN * DDIM;
  const size_t ED = (size_t)NE * DDIM;
  const size_t WD = (size_t)L * DDIM * DDIM;

  // workspace layout
  char* ws = (char*)d_ws;
  auto alignup = [](size_t v) { return (v + 255) & ~(size_t)255; };
  size_t off = 0;
  float* x_buf   = (float*)(ws + off); off = alignup(off + ND * 4);
  float* aggr    = (float*)(ws + off); off = alignup(off + ND * 4);
  bf16*  aggr_bf = (bf16*)(ws + off);  off = alignup(off + ND * 2);
  bf16*  ea_bf   = (bf16*)(ws + off);  off = alignup(off + ED * 2);
  bf16*  we1     = (bf16*)(ws + off);  off = alignup(off + WD * 2);
  bf16*  we2     = (bf16*)(ws + off);  off = alignup(off + WD * 2);
  bf16*  wn1     = (bf16*)(ws + off);  off = alignup(off + WD * 2);
  bf16*  wn2     = (bf16*)(ws + off);  off = alignup(off + WD * 2);

  const float* x0  = (const float*)d_in[0];
  const float* ea  = (const float*)d_in[1];
  const float* e1w = (const float*)d_in[2];
  const float* e1b = (const float*)d_in[3];
  const float* e2w = (const float*)d_in[4];
  const float* e2b = (const float*)d_in[5];
  const float* elg = (const float*)d_in[6];
  const float* elb = (const float*)d_in[7];
  const float* n1w = (const float*)d_in[8];
  const float* n1b = (const float*)d_in[9];
  const float* n2w = (const float*)d_in[10];
  const float* n2b = (const float*)d_in[11];
  const float* nlg = (const float*)d_in[12];
  const float* nlb = (const float*)d_in[13];
  const int*  eidx = (const int*)d_in[14];

  hipMemcpyAsync(x_buf, x0, ND * 4, hipMemcpyDeviceToDevice, stream);
  const int CB = 2048;
  cvt_f32_to_bf16<<<CB, 256, 0, stream>>>(ea,  ea_bf, ED);
  cvt_f32_to_bf16<<<CB, 256, 0, stream>>>(e1w, we1, WD);
  cvt_f32_to_bf16<<<CB, 256, 0, stream>>>(e2w, we2, WD);
  cvt_f32_to_bf16<<<CB, 256, 0, stream>>>(n1w, wn1, WD);
  cvt_f32_to_bf16<<<CB, 256, 0, stream>>>(n2w, wn2, WD);

  const int egrid = (NE + TILE_M - 1) / TILE_M;
  const int ngrid = (NN + TILE_M - 1) / TILE_M;
  const size_t wstep = (size_t)DDIM * DDIM;

  for (int l = 0; l < L; ++l) {
    hipMemsetAsync(aggr, 0, ND * 4, stream);
    edge_layer_kernel<<<egrid, 256, LDS_BYTES, stream>>>(
        ea_bf, we1 + (size_t)l * wstep, e1b + l * DDIM,
        we2 + (size_t)l * wstep, e2b + l * DDIM,
        elg + l * DDIM, elb + l * DDIM,
        x_buf, eidx, eidx + NE, aggr, NE);
    cvt_f32_to_bf16<<<CB, 256, 0, stream>>>(aggr, aggr_bf, ND);
    node_layer_kernel<<<ngrid, 256, LDS_BYTES, stream>>>(
        aggr_bf, wn1 + (size_t)l * wstep, n1b + l * DDIM,
        wn2 + (size_t)l * wstep, n2b + l * DDIM,
        nlg + l * DDIM, nlb + l * DDIM,
        x_buf, NN);
  }
  hipMemcpyAsync(d_out, x_buf, ND * 4, hipMemcpyDeviceToDevice, stream);
}
